// GraphAttentionLayer_39264591020282
// MI455X (gfx1250) — compile-verified, matrix operand order checked by reading, not run on hardware
//
#include <hip/hip_runtime.h>

typedef __attribute__((ext_vector_type(16))) __bf16 v16bf;
typedef __attribute__((ext_vector_type(8)))  __bf16 v8bf;
typedef __attribute__((ext_vector_type(8)))  float  v8f;
typedef __attribute__((ext_vector_type(4)))  float  v4f;

#define N_NODES 8192
#define F_IN    256
#define F_OUT   128
#define ALPHA   0.2f

// ---------------------------------------------------------------------------
// K0: W (256x128 f32, row-major) -> W_t (128x256 bf16) transposed
// ---------------------------------------------------------------------------
__global__ __launch_bounds__(256) void wconv_kernel(const float* __restrict__ W,
                                                    __bf16* __restrict__ Wt) {
    const int n = blockIdx.x;    // 0..127 output feature
    const int k = threadIdx.x;   // 0..255 input feature
    Wt[n * F_IN + k] = (__bf16)W[k * F_OUT + n];
}

// ---------------------------------------------------------------------------
// K1: h = input @ W, stored transposed as bf16 h_t[128][8192].
// One wave computes a 16x128 tile of h with 8 bf16 WMMA accumulators.
// ---------------------------------------------------------------------------
__global__ __launch_bounds__(32) void gemm_h_kernel(const float* __restrict__ input,
                                                    const __bf16* __restrict__ Wt,
                                                    __bf16* __restrict__ ht) {
    const int i0   = blockIdx.x * 16;
    const int lane = threadIdx.x;
    const int r    = lane & 15;          // row within tile (A) / col within frag (B,C)
    const int hi   = lane >> 4;
    const int kb   = hi ? 8 : 0;         // K half per 16-bit A/B VGPR layout

    v8f acc[8] = {};

    for (int kt = 0; kt < F_IN / 32; ++kt) {
        const int k0 = kt * 32;
        // --- A fragment: input rows, K pattern {kb..kb+7, kb+16..kb+23}
        const float* arow = input + (size_t)(i0 + r) * F_IN + k0 + kb;
        v4f a0 = *(const v4f*)(arow);
        v4f a1 = *(const v4f*)(arow + 4);
        v4f a2 = *(const v4f*)(arow + 16);
        v4f a3 = *(const v4f*)(arow + 20);
        v16bf afrag;
#pragma unroll
        for (int t = 0; t < 4; ++t) {
            afrag[t]      = (__bf16)a0[t];
            afrag[4 + t]  = (__bf16)a1[t];
            afrag[8 + t]  = (__bf16)a2[t];
            afrag[12 + t] = (__bf16)a3[t];
        }
#pragma unroll
        for (int f = 0; f < 8; ++f) {
            // --- B fragment: column n = f*16 + r of W, contiguous K runs in W_t
            const __bf16* brow = Wt + (size_t)(f * 16 + r) * F_IN + k0 + kb;
            v8bf b0 = *(const v8bf*)brow;
            v8bf b1 = *(const v8bf*)(brow + 16);
            v16bf bfrag = __builtin_shufflevector(b0, b1, 0, 1, 2, 3, 4, 5, 6, 7,
                                                  8, 9, 10, 11, 12, 13, 14, 15);
            acc[f] = __builtin_amdgcn_wmma_f32_16x16x32_bf16(
                false, afrag, false, bfrag, (short)0, acc[f], false, false);
        }
    }

    // C/D layout: lane holds N = lane&15, M = vgprIdx + 8*(lane>>4).
    // Store transposed bf16: h_t[n][m]
#pragma unroll
    for (int f = 0; f < 8; ++f) {
        const int n = f * 16 + r;
#pragma unroll
        for (int rr = 0; rr < 8; ++rr) {
            const int m = i0 + rr + 8 * hi;
            ht[(size_t)n * N_NODES + m] = (__bf16)acc[f][rr];
        }
    }
}

// ---------------------------------------------------------------------------
// K2: s1[i] = sum_n h[i,n]*a[n]; s2[i] = sum_n h[i,n]*a[128+n]
// Coalesced: consecutive threads read consecutive columns of h_t rows.
// ---------------------------------------------------------------------------
__global__ __launch_bounds__(128) void s_kernel(const __bf16* __restrict__ ht,
                                                const float* __restrict__ a,
                                                float* __restrict__ s1,
                                                float* __restrict__ s2) {
    const int i = blockIdx.x * 128 + threadIdx.x;
    float acc1 = 0.f, acc2 = 0.f;
    for (int n = 0; n < F_OUT; ++n) {
        const float hv = (float)ht[(size_t)n * N_NODES + i];
        acc1 += hv * a[n];
        acc2 += hv * a[F_OUT + n];
    }
    s1[i] = acc1;
    s2[i] = acc2;
}

// ---------------------------------------------------------------------------
// K3: exact masked row max.  leakyrelu is monotonic, so
//     m_i = leakyrelu(s1_i + max_{adj_ij>0} s2_j)
// ---------------------------------------------------------------------------
__global__ __launch_bounds__(256) void rowmax_kernel(const int* __restrict__ adj,
                                                     const float* __restrict__ s1,
                                                     const float* __restrict__ s2,
                                                     float* __restrict__ mrow) {
    __shared__ float red[256];
    const int i   = blockIdx.x;
    const int tid = threadIdx.x;
    float mx = -3.0e38f;
    for (int j = tid; j < N_NODES; j += 256) {
        if (adj[(size_t)i * N_NODES + j] > 0) {
            const float v = s2[j];
            mx = fmaxf(mx, v);
        }
    }
    red[tid] = mx;
    __syncthreads();
    for (int s = 128; s > 0; s >>= 1) {
        if (tid < s) red[tid] = fmaxf(red[tid], red[tid + s]);
        __syncthreads();
    }
    if (tid == 0) {
        const float e = s1[i] + red[0];
        mrow[i] = (e > 0.f) ? e : ALPHA * e;
    }
}

// ---------------------------------------------------------------------------
// K4: fused masked softmax + (P @ h) + ELU.  One wave per 16x128 output tile.
// Streams adj once, probabilities go straight into bf16 WMMA as A fragments;
// h_t provides B fragments as contiguous b128 loads.
// ---------------------------------------------------------------------------
__global__ __launch_bounds__(32) void attn_kernel(const int* __restrict__ adj,
                                                  const __bf16* __restrict__ ht,
                                                  const float* __restrict__ s1,
                                                  const float* __restrict__ s2,
                                                  const float* __restrict__ mrow,
                                                  float* __restrict__ out) {
    __shared__ int   adjT[16 * 32];
    __shared__ float s2T[32];
    __shared__ float denomT[16];

    const int i0   = blockIdx.x * 16;
    const int lane = threadIdx.x;
    const int r    = lane & 15;
    const int hi   = lane >> 4;
    const int kb   = hi ? 8 : 0;

    const float s1r = s1[i0 + r];
    const float mr  = mrow[i0 + r];

    v8f acc[8] = {};
    float denom = 0.f;

    for (int jt = 0; jt < N_NODES / 32; ++jt) {
        const int j0 = jt * 32;

        // Stage adj tile (coalesced: lane = column) and s2 slice through LDS
#pragma unroll
        for (int t = 0; t < 16; ++t)
            adjT[t * 32 + lane] = adj[(size_t)(i0 + t) * N_NODES + j0 + lane];
        s2T[lane] = s2[j0 + lane];
        __syncthreads();

        if (jt + 1 < N_NODES / 32)
            __builtin_prefetch(&adj[(size_t)(i0 + r) * N_NODES + j0 + 32], 0, 1);

        // Build A fragment of probabilities: lane row r, K = kb+{0..7,16..23}
        v16bf p;
#pragma unroll
        for (int t = 0; t < 16; ++t) {
            const int k  = kb + (t & 7) + ((t & 8) << 1);
            const float e  = s1r + s2T[k];
            const float el = (e > 0.f) ? e : ALPHA * e;
            float pv = 0.f;
            if (adjT[r * 32 + k] > 0) pv = __expf(el - mr);
            denom += pv;
            p[t] = (__bf16)pv;
        }
        __syncthreads();

#pragma unroll
        for (int f = 0; f < 8; ++f) {
            const __bf16* brow = ht + (size_t)(f * 16 + r) * N_NODES + j0 + kb;
            v8bf b0 = *(const v8bf*)brow;
            v8bf b1 = *(const v8bf*)(brow + 16);
            v16bf bfrag = __builtin_shufflevector(b0, b1, 0, 1, 2, 3, 4, 5, 6, 7,
                                                  8, 9, 10, 11, 12, 13, 14, 15);
            acc[f] = __builtin_amdgcn_wmma_f32_16x16x32_bf16(
                false, p, false, bfrag, (short)0, acc[f], false, false);
        }
    }

    // Combine per-row denominator across the two K halves (lane <-> lane^16)
    denom += __shfl_xor(denom, 16, 32);
    if (lane < 16) denomT[lane] = denom;
    __syncthreads();

    // Normalize, ELU, store.  Lane holds N = lane&15, M = rr + 8*hi.
#pragma unroll
    for (int f = 0; f < 8; ++f) {
#pragma unroll
        for (int rr = 0; rr < 8; ++rr) {
            const int m = rr + 8 * hi;
            float d = denomT[m];
            d = (d > 1e-20f) ? d : 1e-20f;
            float v = acc[f][rr] / d;
            v = (v > 0.f) ? v : (__expf(v) - 1.f);
            out[(size_t)(i0 + m) * F_OUT + f * 16 + r] = v;
        }
    }
}

// ---------------------------------------------------------------------------
// Launch
// ---------------------------------------------------------------------------
extern "C" void kernel_launch(void* const* d_in, const int* in_sizes, int n_in,
                              void* d_out, int out_size, void* d_ws, size_t ws_size,
                              hipStream_t stream) {
    const float* input = (const float*)d_in[0];   // 8192x256
    const int*   adj   = (const int*)d_in[1];     // 8192x8192
    const float* W     = (const float*)d_in[2];   // 256x128
    const float* a     = (const float*)d_in[3];   // 256
    float*       out   = (float*)d_out;           // 8192x128

    char* ws = (char*)d_ws;
    __bf16* Wt   = (__bf16*)ws;                                      // 64 KB
    __bf16* ht   = (__bf16*)(ws + 0x10000);                          // 2 MB
    float*  s1   = (float*)(ws + 0x10000 + 0x200000);                // 32 KB
    float*  s2   = s1 + N_NODES;                                     // 32 KB
    float*  mrow = s2 + N_NODES;                                     // 32 KB

    wconv_kernel<<<F_OUT, F_IN, 0, stream>>>(W, Wt);
    gemm_h_kernel<<<N_NODES / 16, 32, 0, stream>>>(input, Wt, ht);
    s_kernel<<<N_NODES / 128, 128, 0, stream>>>(ht, a, s1, s2);
    rowmax_kernel<<<N_NODES, 256, 0, stream>>>(adj, s1, s2, mrow);
    attn_kernel<<<N_NODES / 16, 32, 0, stream>>>(adj, ht, s1, s2, mrow, out);
}
